// CustomStridedAttention_57251914056089
// MI455X (gfx1250) — matrix operand order repaired and checked by reading, not compiled, not run on hardware
//
#include <hip/hip_runtime.h>

// ============================================================================
// CustomStridedAttention for MI455X (gfx1250, wave32, WMMA + TDM)
//   x(4,4096,1024) f32 -> QKV proj -> strided attn (keys ::8) -> out proj
// GEMMs via v_wmma_f32_16x16x32_bf16; LDS staging via tensor_load_to_lds
// (Tensor Data Mover, TENSORcnt) with double buffering in the GEMM.
// Workspace (bf16): xb[16.7M] wqkv[3.1M] wout[1M] q[16.7M] ks[2.1M] vT[2.1M]
// attn[16.7M]  (~112 MB).
// ============================================================================

typedef __attribute__((ext_vector_type(16))) __bf16       v16bf;
typedef __attribute__((ext_vector_type(8)))  float        v8f;
typedef __attribute__((ext_vector_type(4)))  unsigned int u32x4;
typedef __attribute__((ext_vector_type(8)))  int          i32x8;
typedef __attribute__((ext_vector_type(4)))  int          i32x4;

#define DEVFN static __device__ __forceinline__

#if defined(__has_builtin)
#if __has_builtin(__builtin_amdgcn_tensor_load_to_lds)
#define HAVE_TDM 1
#endif
#endif
#ifndef HAVE_TDM
#define HAVE_TDM 0
#endif

union FragBF { u32x4 u[2]; v16bf v; };

DEVFN unsigned short f2bf(float f) {
  union { float f; unsigned int u; } x; x.f = f;
  unsigned int r = x.u + 0x7FFFu + ((x.u >> 16) & 1u);   // round-to-nearest-even
  return (unsigned short)(r >> 16);
}

DEVFN v8f zero8() {
  v8f z;
#pragma unroll
  for (int i = 0; i < 8; ++i) z[i] = 0.0f;
  return z;
}

DEVFN v8f wmma_bf16(v16bf a, v16bf b, v8f c) {
  // (neg_a, A, neg_b, B, c_mod, C, reuse_a, reuse_b)
  return __builtin_amdgcn_wmma_f32_16x16x32_bf16(false, a, false, b, (short)0, c,
                                                 false, false);
}

// A fragment 16x32 bf16 (M x K), row-major source, base = &src[row0][k0].
DEVFN v16bf load_a_frag(const unsigned short* base, int lda, int kk, int lane) {
  const unsigned short* p = base + (lane & 15) * lda + kk + ((lane >> 4) << 3);
  FragBF r;
  r.u[0] = *(const u32x4*)(p);
  r.u[1] = *(const u32x4*)(p + 16);
  return r.v;
}

// B fragment 32x16 bf16 (K x N); source is N x K row-major, base = &src[n0][k0].
DEVFN v16bf load_b_frag(const unsigned short* base, int ldb, int kk, int lane) {
  const unsigned short* p = base + (lane & 15) * ldb + kk + ((lane >> 4) << 3);
  FragBF r;
  r.u[0] = *(const u32x4*)(p);
  r.u[1] = *(const u32x4*)(p + 16);
  return r.v;
}

// ---------------------------------------------------------------------------
// Tensor Data Mover: build D# (groups 0/1) per CDNA5 ISA
// and issue tensor_load_to_lds. Issued once per wave (EXEC ignored by TDM).
// ---------------------------------------------------------------------------
#if HAVE_TDM
DEVFN unsigned lds_byte_addr(const void* p) {
  // generic LDS pointer: addr[31:0] is the LDS byte offset (aperture in MSBs)
  return (unsigned)(unsigned long long)p;
}

DEVFN void tdm_load_to_lds(unsigned lds_addr, const void* gptr,
                           unsigned tensor_d0, unsigned tensor_d1,
                           unsigned tile_d0, unsigned tile_d1,
                           unsigned stride0_elems, int pad_interval_code,
                           int pad_amount_code) {
  const unsigned long long ga = (unsigned long long)gptr;
  u32x4 g0;
  g0[0] = 1u;                                        // count=1, user load D#
  g0[1] = lds_addr;                                  // LDS byte address
  g0[2] = (unsigned)ga;                              // global_addr[31:0]
  g0[3] = (unsigned)((ga >> 32) & 0x01FFFFFFu) | (2u << 30);  // [56:32], type=2
  unsigned w0 = (1u << 16);                          // data_size=1 -> 2 bytes
  if (pad_amount_code >= 0)
    w0 |= (1u << 20) | ((unsigned)pad_interval_code << 22) |
          ((unsigned)pad_amount_code << 25);         // pad_enable/interval/amt
  i32x8 g1;
  g1[0] = (int)w0;
  g1[1] = (int)((tensor_d0 & 0xFFFFu) << 16);        // tensor_dim0[15:0]
  g1[2] = (int)((tensor_d0 >> 16) | ((tensor_d1 & 0xFFFFu) << 16));
  g1[3] = (int)((tensor_d1 >> 16) | ((tile_d0 & 0xFFFFu) << 16));
  g1[4] = (int)(tile_d1 & 0xFFFFu);                  // tile_dim1 (tile_dim2=0)
  g1[5] = (int)stride0_elems;                        // tensor_dim0_stride[31:0]
  g1[6] = 0;
  g1[7] = 0;
  i32x4 z4; z4[0] = z4[1] = z4[2] = z4[3] = 0;
#if defined(__clang_major__) && (__clang_major__ >= 23)
  i32x8 z8;
#pragma unroll
  for (int i = 0; i < 8; ++i) z8[i] = 0;
  __builtin_amdgcn_tensor_load_to_lds(g0, g1, z4, z4, z8, 0);
#else
  __builtin_amdgcn_tensor_load_to_lds(g0, g1, z4, z4, 0);
#endif
}

// 128x64 bf16 tile of a row-major [rows x ld] matrix, padded to 80-ushort pitch
DEVFN void tdm_tile_128x64(const unsigned short* lds, const unsigned short* g,
                           int ld) {
  // pad_interval code 4 = 32 DWORDs (one 64-elem row), pad code 7 = 8 DWORDs
  tdm_load_to_lds(lds_byte_addr(lds), g, 1u << 20, 1u << 20, 64u, 128u,
                  (unsigned)ld, 4, 7);
}

// contiguous n-element bf16 copy (n <= 65535)
DEVFN void tdm_linear(const unsigned short* lds, const unsigned short* g,
                      unsigned n) {
  tdm_load_to_lds(lds_byte_addr(lds), g, n, 1u, n, 1u, n, 0, -1);
}
#endif  // HAVE_TDM

// ---------------------------------------------------------------------------
// Kernel 1: fp32 -> bf16 conversion (n multiple of 256)
// ---------------------------------------------------------------------------
__global__ __launch_bounds__(256) void cvt_f32_bf16(const float* __restrict__ in,
                                                    unsigned short* __restrict__ out,
                                                    int n) {
  int i = blockIdx.x * 256 + threadIdx.x;
  if (i < n) out[i] = f2bf(in[i]);
}

// ---------------------------------------------------------------------------
// Kernel 2/4: C[M,N] = A[M,K] @ W[N,K]^T + bias, bf16 in, f32 accum.
// Block tile 128x128, 8 waves, wave tile 64x32. TDM double-buffered staging.
// EPI==0: scatter into q (x0.125), strided ks, strided+transposed vT.
// EPI==1: fp32 row-major output.
// ---------------------------------------------------------------------------
template <int EPI>
__global__ __launch_bounds__(256) void gemm_bf16_tn(
    const unsigned short* __restrict__ A, const unsigned short* __restrict__ W,
    const float* __restrict__ bias, int Kdim,
    unsigned short* __restrict__ qb, unsigned short* __restrict__ ksb,
    unsigned short* __restrict__ vtb, float* __restrict__ outb) {
  const int LDT = 80;  // 64 + 16 pad (16B-aligned rows, bank spread)
  __shared__ unsigned short As[2][128 * LDT];
  __shared__ unsigned short Bs[2][128 * LDT];

  const int tid  = threadIdx.x;
  const int lane = tid & 31;
  const int wid  = tid >> 5;
  const int wm   = (wid >> 2) * 64;
  const int wn   = (wid & 3) * 32;
  const int gm0  = blockIdx.y * 128;
  const int gn0  = blockIdx.x * 128;

  const unsigned short* Arow = A + (size_t)gm0 * Kdim;
  const unsigned short* Wrow = W + (size_t)gn0 * Kdim;

  v8f acc[4][2];
#pragma unroll
  for (int mt = 0; mt < 4; ++mt)
#pragma unroll
    for (int nt = 0; nt < 2; ++nt) acc[mt][nt] = zero8();

#if HAVE_TDM
  if (wid == 0) {  // prologue: DMA first tiles into buffer 0
    tdm_tile_128x64(&As[0][0], Arow, Kdim);
    tdm_tile_128x64(&Bs[0][0], Wrow, Kdim);
  }
#endif

  for (int kt = 0; kt < Kdim; kt += 64) {
    const int pb = (kt >> 6) & 1;
#if HAVE_TDM
    if (wid == 0) {
      if (kt + 64 < Kdim) {  // DMA next tiles into other buffer, then make
        tdm_tile_128x64(&As[pb ^ 1][0], Arow + kt + 64, Kdim);
        tdm_tile_128x64(&Bs[pb ^ 1][0], Wrow + kt + 64, Kdim);
        __builtin_amdgcn_s_wait_tensorcnt(2);  // current pair complete
      } else {
        __builtin_amdgcn_s_wait_tensorcnt(0);
      }
    }
#else
#pragma unroll
    for (int i = 0; i < 4; ++i) {
      int lin = tid + i * 256;
      int row = lin >> 3;
      int c8  = (lin & 7) << 3;
      *(u32x4*)&As[pb][row * LDT + c8] =
          *(const u32x4*)&Arow[(size_t)row * Kdim + kt + c8];
      *(u32x4*)&Bs[pb][row * LDT + c8] =
          *(const u32x4*)&Wrow[(size_t)row * Kdim + kt + c8];
    }
#endif
    if (kt + 64 < Kdim) {  // L2 hint for the next tile: global_prefetch_b8
      __builtin_prefetch(&Arow[(size_t)(tid >> 1) * Kdim + kt + 64 + (tid & 1) * 32], 0, 1);
      __builtin_prefetch(&Wrow[(size_t)(tid >> 1) * Kdim + kt + 64 + (tid & 1) * 32], 0, 1);
    }
    __syncthreads();

#pragma unroll
    for (int kk = 0; kk < 64; kk += 32) {
      v16bf af[4], bfr[2];
#pragma unroll
      for (int mt = 0; mt < 4; ++mt)
        af[mt] = load_a_frag(&As[pb][(wm + mt * 16) * LDT], LDT, kk, lane);
#pragma unroll
      for (int nt = 0; nt < 2; ++nt)
        bfr[nt] = load_b_frag(&Bs[pb][(wn + nt * 16) * LDT], LDT, kk, lane);
#pragma unroll
      for (int mt = 0; mt < 4; ++mt)
#pragma unroll
        for (int nt = 0; nt < 2; ++nt)
          acc[mt][nt] = wmma_bf16(af[mt], bfr[nt], acc[mt][nt]);
    }
    __syncthreads();
  }

  // Epilogue. C-fragment: lane&15 = column; VGPR g = row g (+8 on upper half).
  const int lh = lane >> 4;
  const int ln = lane & 15;
#pragma unroll
  for (int mt = 0; mt < 4; ++mt) {
#pragma unroll
    for (int nt = 0; nt < 2; ++nt) {
      const int c    = gn0 + wn + nt * 16 + ln;
      const float bv = bias[c];
#pragma unroll
      for (int g = 0; g < 8; ++g) {
        const int r   = gm0 + wm + mt * 16 + g + lh * 8;
        const float v = acc[mt][nt][g] + bv;
        if (EPI == 0) {
          const int t = c >> 10, e = c & 1023;
          const int h = e >> 6, d = e & 63;
          const int b = r >> 12, s = r & 4095;
          if (t == 0) {  // Q, pre-scaled by 1/sqrt(64)
            qb[((size_t)((b * 16 + h) * 4096 + s)) * 64 + d] = f2bf(v * 0.125f);
          } else if (t == 1) {  // K, every 8th position
            if ((s & 7) == 0)
              ksb[((size_t)((b * 16 + h) * 512 + (s >> 3))) * 64 + d] = f2bf(v);
          } else {  // V, strided + transposed: (b,h,d,sk)
            if ((s & 7) == 0)
              vtb[((size_t)((b * 16 + h) * 64 + d)) * 512 + (s >> 3)] = f2bf(v);
          }
        } else {
          outb[(size_t)r * 1024 + c] = v;
        }
      }
    }
  }
}

// ---------------------------------------------------------------------------
// Kernel 3: flash attention over 512 strided keys. One block = one (b,h) x
// 128 queries; 8 waves x 16-query tiles. K (64KB), V^T (64KB) and Q chunk
// (16KB) staged via TDM; per-wave 1KB LDS buffer for probs C->A transpose.
// LDS total 152 KB (<320 KB/WGP).
// ---------------------------------------------------------------------------
__global__ __launch_bounds__(256) void attn_kernel(
    const unsigned short* __restrict__ q, const unsigned short* __restrict__ ks,
    const unsigned short* __restrict__ vt, unsigned short* __restrict__ attn) {
  __shared__ unsigned short ks_s[512 * 64];    // 64 KB
  __shared__ unsigned short vt_s[64 * 512];    // 64 KB
  __shared__ unsigned short q_s[128 * 64];     // 16 KB
  __shared__ unsigned short pw_s[8][16 * 32];  // 8 KB

  const int tid  = threadIdx.x;
  const int lane = tid & 31;
  const int wid  = tid >> 5;
  const int bh   = blockIdx.x >> 5;
  const int qc   = blockIdx.x & 31;
  const size_t bh_off = (size_t)bh * 512 * 64;
  const size_t q_off  = ((size_t)bh * 4096 + qc * 128) * 64;

#if HAVE_TDM
  if (wid == 0) {
    tdm_linear(ks_s, ks + bh_off, 32768u);
    tdm_linear(vt_s, vt + bh_off, 32768u);
    tdm_linear(q_s, q + q_off, 8192u);
    __builtin_amdgcn_s_wait_tensorcnt(0);
  }
#else
#pragma unroll
  for (int i = 0; i < 16; ++i) {
    const int lin = (tid + i * 256) * 8;
    *(u32x4*)&ks_s[lin] = *(const u32x4*)&ks[bh_off + lin];
    *(u32x4*)&vt_s[lin] = *(const u32x4*)&vt[bh_off + lin];
  }
#pragma unroll
  for (int i = 0; i < 4; ++i) {
    const int lin = (tid + i * 256) * 8;
    *(u32x4*)&q_s[lin] = *(const u32x4*)&q[q_off + lin];
  }
#endif
  __syncthreads();

  const v16bf aq0 = load_a_frag(&q_s[(wid * 16) * 64], 64, 0, lane);   // d 0..31
  const v16bf aq1 = load_a_frag(&q_s[(wid * 16) * 64], 64, 32, lane);  // d 32..63

  v8f oacc[4];
#pragma unroll
  for (int nt = 0; nt < 4; ++nt) oacc[nt] = zero8();
  float mrow[8], lrow[8];
#pragma unroll
  for (int g = 0; g < 8; ++g) { mrow[g] = -1e30f; lrow[g] = 0.0f; }

  unsigned short* pw = &pw_s[wid][0];
  const int lh = lane >> 4;
  const int ln = lane & 15;

  for (int kt = 0; kt < 512; kt += 32) {
    // scores S = Q (16x64) @ K^T (64x32): two 16x16 tiles, 2 chained WMMA each
    v8f s0 = wmma_bf16(aq0, load_b_frag(&ks_s[(kt + 0) * 64], 64, 0, lane), zero8());
    s0     = wmma_bf16(aq1, load_b_frag(&ks_s[(kt + 0) * 64], 64, 32, lane), s0);
    v8f s1 = wmma_bf16(aq0, load_b_frag(&ks_s[(kt + 16) * 64], 64, 0, lane), zero8());
    s1     = wmma_bf16(aq1, load_b_frag(&ks_s[(kt + 16) * 64], 64, 32, lane), s1);

    // per-row chunk max across the 16 lanes of each half-wave
    float cm[8];
#pragma unroll
    for (int g = 0; g < 8; ++g) cm[g] = fmaxf(s0[g], s1[g]);
#pragma unroll
    for (int off = 1; off < 16; off <<= 1)
#pragma unroll
      for (int g = 0; g < 8; ++g) cm[g] = fmaxf(cm[g], __shfl_xor(cm[g], off, 32));

    float p0[8], p1[8];
#pragma unroll
    for (int g = 0; g < 8; ++g) {
      const float mn   = fmaxf(mrow[g], cm[g]);
      const float corr = __expf(mrow[g] - mn);
      mrow[g] = mn;
      p0[g]   = __expf(s0[g] - mn);
      p1[g]   = __expf(s1[g] - mn);
      lrow[g] = lrow[g] * corr + p0[g] + p1[g];  // per-lane partial row sum
#pragma unroll
      for (int nt = 0; nt < 4; ++nt) oacc[nt][g] *= corr;
    }

    // C-layout -> A-layout via per-wave LDS round-trip
#pragma unroll
    for (int g = 0; g < 8; ++g) {
      const int m = g + lh * 8;
      pw[m * 32 + ln]      = f2bf(p0[g]);
      pw[m * 32 + 16 + ln] = f2bf(p1[g]);
    }
    asm volatile("s_wait_dscnt 0" ::: "memory");
    const v16bf ap = load_a_frag(pw, 32, 0, lane);  // probs 16x32

    // O += P (16x32) @ V (32x64): 4 WMMAs against V^T columns
#pragma unroll
    for (int nt = 0; nt < 4; ++nt)
      oacc[nt] = wmma_bf16(ap, load_b_frag(&vt_s[(nt * 16) * 512 + kt], 512, 0, lane),
                           oacc[nt]);
  }

  // finalize: reduce row sums, normalize, store (b,s,h,d) bf16
#pragma unroll
  for (int off = 1; off < 16; off <<= 1)
#pragma unroll
    for (int g = 0; g < 8; ++g) lrow[g] += __shfl_xor(lrow[g], off, 32);
  float inv[8];
#pragma unroll
  for (int g = 0; g < 8; ++g) inv[g] = 1.0f / lrow[g];

  const int b = bh >> 4, h = bh & 15;
  const int qbase = qc * 128 + wid * 16;
#pragma unroll
  for (int nt = 0; nt < 4; ++nt) {
    const int e = h * 64 + nt * 16 + ln;
#pragma unroll
    for (int g = 0; g < 8; ++g) {
      const int srow = qbase + g + lh * 8;
      attn[((size_t)(b * 4096 + srow)) * 1024 + e] = f2bf(oacc[nt][g] * inv[g]);
    }
  }
}

// ---------------------------------------------------------------------------
// Host launcher
// ---------------------------------------------------------------------------
extern "C" void kernel_launch(void* const* d_in, const int* in_sizes, int n_in,
                              void* d_out, int out_size, void* d_ws, size_t ws_size,
                              hipStream_t stream) {
  (void)in_sizes; (void)n_in; (void)out_size; (void)ws_size;
  const float* x     = (const float*)d_in[0];
  const float* qkv_w = (const float*)d_in[1];
  const float* qkv_b = (const float*)d_in[2];
  const float* out_w = (const float*)d_in[3];
  const float* out_b = (const float*)d_in[4];
  float* out = (float*)d_out;

  const int E = 1024;
  const size_t M = (size_t)4 * 4096;  // B*S

  unsigned short* ws   = (unsigned short*)d_ws;
  unsigned short* xb   = ws;                          // M*E
  unsigned short* wqkv = xb + M * E;                  // 3E*E
  unsigned short* wout = wqkv + (size_t)3 * E * E;    // E*E
  unsigned short* qb   = wout + (size_t)E * E;        // M*E
  unsigned short* ksb  = qb + M * E;                  // 64*512*64
  unsigned short* vtb  = ksb + (size_t)64 * 512 * 64; // 64*64*512
  unsigned short* at   = vtb + (size_t)64 * 512 * 64; // M*E

  cvt_f32_bf16<<<(int)(M * E / 256), 256, 0, stream>>>(x, xb, (int)(M * E));
  cvt_f32_bf16<<<3 * E * E / 256, 256, 0, stream>>>(qkv_w, wqkv, 3 * E * E);
  cvt_f32_bf16<<<E * E / 256, 256, 0, stream>>>(out_w, wout, E * E);

  gemm_bf16_tn<0><<<dim3(3 * E / 128, (int)(M / 128)), 256, 0, stream>>>(
      xb, wqkv, qkv_b, E, qb, ksb, vtb, nullptr);

  attn_kernel<<<dim3(64 * 32), 256, 0, stream>>>(qb, ksb, vtb, at);

  gemm_bf16_tn<1><<<dim3(E / 128, (int)(M / 128)), 256, 0, stream>>>(
      at, wout, out_b, E, nullptr, nullptr, nullptr, out);
}
